// LayerAdapter_56959856279775
// MI455X (gfx1250) — matrix operand order repaired and checked by reading, not compiled, not run on hardware
//
#include <hip/hip_runtime.h>
#include <hip/hip_bf16.h>

// ---------------------------------------------------------------------------
// Problem constants (B,T,D,L,H) = (4,1024,768,13,12), dh=64
// ---------------------------------------------------------------------------
#define BQ 4
#define TQ 1024
#define DQ 768
#define LQ 13
#define HQ 12
#define BT (BQ*TQ)        /* 4096 tokens */
#define DH 64             /* head dim    */

typedef __bf16 bf16_t;
typedef __attribute__((ext_vector_type(16))) __bf16 v16bf;
typedef __attribute__((ext_vector_type(8)))  float  v8f;

// ---------------------------------------------------------------------------
// CDNA5 async global->LDS copy (ASYNCcnt-tracked, bypasses VGPRs).
// Per CDNA5 ISA: each lane copies 16B from its global address to its LDS offset.
// Generic LDS pointers carry the DS offset in their low 32 bits.
// ---------------------------------------------------------------------------
__device__ __forceinline__ void async_load_b128(void* lds, const void* gaddr) {
  unsigned int loff = (unsigned int)(unsigned long long)lds;
  asm volatile("global_load_async_to_lds_b128 %0, %1, off"
               :: "v"(loff), "v"(gaddr)
               : "memory");
}
__device__ __forceinline__ void wait_async0() {
  asm volatile("s_wait_asynccnt 0x0" ::: "memory");
}

// ---------------------------------------------------------------------------
// 1) inputs (B,T,D,L) f32  ->  Xbf (L, BT, D) bf16
//    each thread reads 13 contiguous floats (52B), writes 13 strided bf16
// ---------------------------------------------------------------------------
__global__ void k_convert_inputs(const float* __restrict__ x,
                                 bf16_t* __restrict__ xbf) {
  size_t gid = (size_t)blockIdx.x * blockDim.x + threadIdx.x;
  if (gid >= (size_t)BT * DQ) return;
  size_t bt = gid / DQ, d = gid % DQ;
  const float* p = x + (bt * DQ + d) * LQ;
#pragma unroll
  for (int l = 0; l < LQ; ++l)
    xbf[(size_t)l * BT * DQ + bt * DQ + d] = (bf16_t)p[l];
}

// ---------------------------------------------------------------------------
// 2) plain f32 -> bf16 convert (for Wq/Wk/Wv/Wo)
// ---------------------------------------------------------------------------
__global__ void k_convert_f32_bf16(const float* __restrict__ s,
                                   bf16_t* __restrict__ d, int n) {
  int gid = blockIdx.x * blockDim.x + threadIdx.x;
  if (gid < n) d[gid] = (bf16_t)s[gid];
}

// ---------------------------------------------------------------------------
// 3) layer_W (l,e,d) f32 -> lWT (l,d,e) bf16  (LDS tile transpose)
// ---------------------------------------------------------------------------
__global__ void k_transpose_layerW(const float* __restrict__ w,
                                   bf16_t* __restrict__ wt) {
  __shared__ float tile[32][33];
  int l  = blockIdx.z;
  int e0 = blockIdx.y * 32, d0 = blockIdx.x * 32;
  int tx = threadIdx.x, ty = threadIdx.y;      // 32 x 8 threads
  const float* src = w + (size_t)l * DQ * DQ;
#pragma unroll
  for (int j = 0; j < 4; ++j)
    tile[ty + j * 8][tx] = src[(size_t)(e0 + ty + j * 8) * DQ + d0 + tx];
  __syncthreads();
  bf16_t* dst = wt + (size_t)l * DQ * DQ;
#pragma unroll
  for (int j = 0; j < 4; ++j)
    dst[(size_t)(d0 + ty + j * 8) * DQ + e0 + tx] = (bf16_t)tile[tx][ty + j * 8];
}

// ---------------------------------------------------------------------------
// 4) combined v-bias: bc[l,f] = sum_e Wv[f,e] * layer_b[l,e] + bv[f]
// ---------------------------------------------------------------------------
__global__ void k_combine_bias(const float* __restrict__ Wv,
                               const float* __restrict__ lb,
                               const float* __restrict__ bv,
                               float* __restrict__ bc) {
  int gid = blockIdx.x * blockDim.x + threadIdx.x;
  if (gid >= LQ * DQ) return;
  int l = gid / DQ, f = gid % DQ;
  const float* wrow = Wv + (size_t)f * DQ;
  const float* brow = lb + (size_t)l * DQ;
  float s = 0.f;
  for (int e = 0; e < DQ; ++e) s += wrow[e] * brow[e];
  bc[gid] = s + bv[f];
}

// ---------------------------------------------------------------------------
// 5) Generic bf16 WMMA GEMM:  C(M,N) = A(M,K) @ W(N,K)^T (+ bias[N])
//    Block tile 128x128x64, 256 threads (8 waves = 4x2 wave grid),
//    each wave owns a 32x64 tile = 2x4 v_wmma_f32_16x16x32_bf16 tiles.
//    Double-buffered LDS filled by global_load_async_to_lds_b128; one
//    barrier per k-step. gridDim.z batches independent GEMM instances.
// ---------------------------------------------------------------------------
#define BLK_M 128
#define BLK_N 128
#define BLK_K 64
#define LDSW  72     /* padded LDS row pitch in bf16 elems (144B) */

union ABfrag { v16bf v; uint4 u[2]; };

template <bool OUT_BF16, bool MASK, bool HAS_BIAS>
__global__ __launch_bounds__(256, 2) void k_gemm(
    const bf16_t* __restrict__ A, long long aBatch,
    const bf16_t* __restrict__ W, long long wBatch,
    const float*  __restrict__ bias, long long biasBatch,
    void* __restrict__ Cv, long long cBatch,
    int M, int N, int K,
    const unsigned char* __restrict__ mask) {
  __shared__ __align__(16) bf16_t As[2][BLK_M * LDSW];
  __shared__ __align__(16) bf16_t Ws[2][BLK_N * LDSW];

  int z = blockIdx.z;
  A += (size_t)z * aBatch;
  W += (size_t)z * wBatch;
  if (HAS_BIAS) bias += (size_t)z * biasBatch;

  int m0 = blockIdx.y * BLK_M, n0 = blockIdx.x * BLK_N;
  int tid  = threadIdx.x;
  int wave = tid >> 5, lane = tid & 31;
  int wm = wave >> 1, wn = wave & 1;      // 4 waves on M x 2 waves on N
  int half = lane >> 4, lrow = lane & 15;

  v8f acc[2][4];
#pragma unroll
  for (int i = 0; i < 2; ++i)
#pragma unroll
    for (int j = 0; j < 4; ++j)
      acc[i][j] = (v8f){0.f, 0.f, 0.f, 0.f, 0.f, 0.f, 0.f, 0.f};

  // issue the 8 async 16B-chunk copies for one (A,W) k-tile into buffer `buf`
  auto issue_tile = [&](int buf, int k0) {
#pragma unroll
    for (int i = 0; i < 4; ++i) {
      int c = tid + i * 256;
      int r = c >> 3, cc = (c & 7) * 8;            // 128 rows x 8 chunks
      async_load_b128(&As[buf][r * LDSW + cc],
                      A + (size_t)(m0 + r) * K + k0 + cc);
      async_load_b128(&Ws[buf][r * LDSW + cc],
                      W + (size_t)(n0 + r) * K + k0 + cc);
    }
  };

  int kTiles = K / BLK_K;
  issue_tile(0, 0);

  for (int kt = 0; kt < kTiles; ++kt) {
    int cur = kt & 1;
    wait_async0();        // my async copies into `cur` are done
    __syncthreads();      // everyone's are done; prev buffer fully consumed
    if (kt + 1 < kTiles) issue_tile(cur ^ 1, (kt + 1) * BLK_K);

#pragma unroll
    for (int kk = 0; kk < BLK_K; kk += 32) {
      ABfrag a[2], b[4];
      // A fragment (16x32 bf16): lane<16 -> K kk+[0..7],kk+[16..23];
      //                          lane>=16 -> +8 on both chunks
#pragma unroll
      for (int mt = 0; mt < 2; ++mt) {
        int r = wm * 32 + mt * 16 + lrow;
        const bf16_t* base = &As[cur][r * LDSW + kk + half * 8];
        a[mt].u[0] = *reinterpret_cast<const uint4*>(base);
        a[mt].u[1] = *reinterpret_cast<const uint4*>(base + 16);
      }
      // B fragment (32x16): column n = lrow; lane<16 -> K kk+[0..7],kk+[16..23]; lane>=16 -> +8
#pragma unroll
      for (int nt = 0; nt < 4; ++nt) {
        int r = wn * 64 + nt * 16 + lrow;
        const bf16_t* base = &Ws[cur][r * LDSW + kk + half * 8];
        b[nt].u[0] = *reinterpret_cast<const uint4*>(base);
        b[nt].u[1] = *reinterpret_cast<const uint4*>(base + 16);
      }
#pragma unroll
      for (int mt = 0; mt < 2; ++mt)
#pragma unroll
        for (int nt = 0; nt < 4; ++nt)
          acc[mt][nt] = __builtin_amdgcn_wmma_f32_16x16x32_bf16(
              false, a[mt].v, false, b[nt].v, (short)0, acc[mt][nt], false, false);
    }
  }

  // epilogue: C layout lane<16: row=j col=lane ; lane>=16: row=8+j col=lane-16
#pragma unroll
  for (int nt = 0; nt < 4; ++nt) {
    int gn = n0 + wn * 64 + nt * 16 + lrow;
    float bb = HAS_BIAS ? bias[gn] : 0.f;
#pragma unroll
    for (int mt = 0; mt < 2; ++mt) {
#pragma unroll
      for (int j = 0; j < 8; ++j) {
        int gm = m0 + wm * 32 + mt * 16 + half * 8 + j;
        float val = acc[mt][nt][j] + bb;
        if (MASK) {
          if (mask[gm]) val = 0.f;
        }
        if (OUT_BF16)
          ((bf16_t*)Cv)[(size_t)z * cBatch + (size_t)gm * N + gn] = (bf16_t)val;
        else
          ((float*)Cv)[(size_t)z * cBatch + (size_t)gm * N + gn] = val;
      }
    }
  }
}

// ---------------------------------------------------------------------------
// 6) one-query attention over L=13 layer states; thread = (token, head)
// ---------------------------------------------------------------------------
__global__ void k_attn(const bf16_t* __restrict__ q,
                       const bf16_t* __restrict__ kmat,
                       const bf16_t* __restrict__ vmat,
                       bf16_t* __restrict__ attn) {
  int gid = blockIdx.x * blockDim.x + threadIdx.x;
  if (gid >= BT * HQ) return;
  int t = gid / HQ, h = gid % HQ;

  const bf16_t* qp = q + (size_t)t * DQ + h * DH;
  float qr[DH];
#pragma unroll
  for (int i = 0; i < DH; ++i) qr[i] = (float)qp[i];

  float sc[LQ];
  float mx = -1e30f;
  for (int l = 0; l < LQ; ++l) {
    const bf16_t* kp = kmat + ((size_t)l * BT + t) * DQ + h * DH;
    float s = 0.f;
#pragma unroll
    for (int i = 0; i < DH; ++i) s += qr[i] * (float)kp[i];
    s *= 0.125f;                       /* 1/sqrt(64) */
    sc[l] = s;
    mx = fmaxf(mx, s);
  }
  float den = 0.f;
  for (int l = 0; l < LQ; ++l) {
    float e = __expf(sc[l] - mx);
    sc[l] = e;
    den += e;
  }
  float inv = 1.f / den;

  float ac[DH];
#pragma unroll
  for (int i = 0; i < DH; ++i) ac[i] = 0.f;
  for (int l = 0; l < LQ; ++l) {
    const bf16_t* vp = vmat + ((size_t)l * BT + t) * DQ + h * DH;
    float w = sc[l] * inv;
#pragma unroll
    for (int i = 0; i < DH; ++i) ac[i] += w * (float)vp[i];
  }
  bf16_t* op = attn + (size_t)t * DQ + h * DH;
#pragma unroll
  for (int i = 0; i < DH; ++i) op[i] = (bf16_t)ac[i];
}

// ---------------------------------------------------------------------------
// host launcher
// ---------------------------------------------------------------------------
extern "C" void kernel_launch(void* const* d_in, const int* in_sizes, int n_in,
                              void* d_out, int out_size, void* d_ws,
                              size_t ws_size, hipStream_t stream) {
  const float*         inputs  = (const float*)d_in[0];
  const unsigned char* mask    = (const unsigned char*)d_in[1];
  const float*         layer_W = (const float*)d_in[2];
  const float*         layer_b = (const float*)d_in[3];
  const float* Wq = (const float*)d_in[4];  const float* bq = (const float*)d_in[5];
  const float* Wk = (const float*)d_in[6];  const float* bk = (const float*)d_in[7];
  const float* Wv = (const float*)d_in[8];  const float* bv = (const float*)d_in[9];
  const float* Wo = (const float*)d_in[10]; const float* bo = (const float*)d_in[11];
  float* out = (float*)d_out;

  // workspace carve-out (256B aligned)
  char* ws = (char*)d_ws;
  size_t off = 0;
  auto carve = [&](size_t bytes) -> void* {
    void* p = ws + off;
    off += (bytes + 255) & ~(size_t)255;
    return p;
  };
  bf16_t* Xbf    = (bf16_t*)carve((size_t)LQ * BT * DQ * 2);
  bf16_t* Wqb    = (bf16_t*)carve((size_t)DQ * DQ * 2);
  bf16_t* Wkb    = (bf16_t*)carve((size_t)DQ * DQ * 2);
  bf16_t* Wvb    = (bf16_t*)carve((size_t)DQ * DQ * 2);
  bf16_t* Wob    = (bf16_t*)carve((size_t)DQ * DQ * 2);
  bf16_t* lWT    = (bf16_t*)carve((size_t)LQ * DQ * DQ * 2);
  bf16_t* Wc     = (bf16_t*)carve((size_t)LQ * DQ * DQ * 2);
  float*  bc     = (float*)carve((size_t)LQ * DQ * 4);
  bf16_t* qbf    = (bf16_t*)carve((size_t)BT * DQ * 2);
  bf16_t* kbf    = (bf16_t*)carve((size_t)LQ * BT * DQ * 2);
  bf16_t* vbf    = (bf16_t*)carve((size_t)LQ * BT * DQ * 2);
  bf16_t* attnbf = (bf16_t*)carve((size_t)BT * DQ * 2);

  // --- stage 1: precision conversion / layout transforms ---
  k_convert_inputs<<<(BT * DQ + 255) / 256, 256, 0, stream>>>(inputs, Xbf);
  k_convert_f32_bf16<<<(DQ * DQ + 255) / 256, 256, 0, stream>>>(Wq, Wqb, DQ * DQ);
  k_convert_f32_bf16<<<(DQ * DQ + 255) / 256, 256, 0, stream>>>(Wk, Wkb, DQ * DQ);
  k_convert_f32_bf16<<<(DQ * DQ + 255) / 256, 256, 0, stream>>>(Wv, Wvb, DQ * DQ);
  k_convert_f32_bf16<<<(DQ * DQ + 255) / 256, 256, 0, stream>>>(Wo, Wob, DQ * DQ);
  k_transpose_layerW<<<dim3(24, 24, LQ), dim3(32, 8), 0, stream>>>(layer_W, lWT);
  k_combine_bias<<<(LQ * DQ + 255) / 256, 256, 0, stream>>>(Wv, layer_b, bv, bc);

  // --- stage 2: fold Wv into each layer_W:  Wc[l] = Wv @ layer_W[l]  ---
  k_gemm<true, false, false><<<dim3(6, 6, LQ), 256, 0, stream>>>(
      Wvb, 0, lWT, (long long)DQ * DQ, nullptr, 0,
      Wc, (long long)DQ * DQ, DQ, DQ, DQ, nullptr);

  // --- stage 3: q / k / v projections (all WMMA) ---
  k_gemm<true, false, true><<<dim3(6, 32, 1), 256, 0, stream>>>(
      Xbf + (size_t)(LQ - 1) * BT * DQ, 0, Wqb, 0, bq, 0,
      qbf, 0, BT, DQ, DQ, nullptr);
  k_gemm<true, false, true><<<dim3(6, 32, LQ), 256, 0, stream>>>(
      Xbf, (long long)BT * DQ, Wkb, 0, bk, 0,
      kbf, (long long)BT * DQ, BT, DQ, DQ, nullptr);
  k_gemm<true, false, true><<<dim3(6, 32, LQ), 256, 0, stream>>>(
      Xbf, (long long)BT * DQ, Wc, (long long)DQ * DQ, bc, DQ,
      vbf, (long long)BT * DQ, BT, DQ, DQ, nullptr);

  // --- stage 4: softmax attention over the 13 layer states ---
  k_attn<<<(BT * HQ + 255) / 256, 256, 0, stream>>>(qbf, kbf, vbf, attnbf);

  // --- stage 5: output projection + key-padding mask (f32 out) ---
  k_gemm<false, true, true><<<dim3(6, 32, 1), 256, 0, stream>>>(
      attnbf, 0, Wob, 0, bo, 0,
      out, 0, BT, DQ, DQ, mask);
}
